// i_vec_layer_25615184953854
// MI455X (gfx1250) — compile-verified, matrix operand order checked by reading, not run on hardware
//
#include <hip/hip_runtime.h>
#include <math.h>
#include <stdint.h>

#define TV   600
#define RP   640        // TV padded to 5x128 (pad rows of G are zero; +I makes pad block identity)
#define ND   60
#define NM   2048
#define NF   10000
#define TP   10240      // frames padded to multiple of 256
#define NK   122880     // ND*NM (mix-major flatten)
#define NKP  (NK / 2)   // packed bf16 k-pairs per row
#define KF   128        // feature K: 60 quadratic + 60 linear + 1 bias + 7 zero
#define NT   (RP / 128) // 5 SYRK tile rows/cols
#define KSPLIT 16       // SYRK K-reduction slices

typedef __attribute__((ext_vector_type(16))) __bf16 v16bf;
typedef __attribute__((ext_vector_type(8)))  float  v8f;

// ---------------------------------------------------------------- helpers ---

__device__ inline v8f v8zero() {
  v8f z;
#pragma unroll
  for (int i = 0; i < 8; ++i) z[i] = 0.f;
  return z;
}

// split two f32 into packed bf16 hi/lo pairs (element k even in [15:0], odd in [31:16])
__device__ inline void split2(float a, float b, unsigned& hi, unsigned& lo) {
  __bf16 ah = (__bf16)a, bh = (__bf16)b;
  __bf16 al = (__bf16)(a - (float)ah), bl = (__bf16)(b - (float)bh);
  union { __bf16 h[2]; unsigned u; } ph, pl;
  ph.h[0] = ah; ph.h[1] = bh;
  pl.h[0] = al; pl.h[1] = bl;
  hi = ph.u; lo = pl.u;
}

// Build one 16x32 bf16 WMMA operand fragment from an LDS tile of packed k-pairs.
// Tile layout: T[row][kpair], 16 kpairs = 32 K values per chunk.
// Lane (m or n) = lane&15, lg = lane>>4. K-pair index for reg p: (p>>2)*8 + lg*4 + (p&3).
__device__ inline v16bf ldfrag(const unsigned T[][16], int row, int lg) {
  union { unsigned u[8]; v16bf v; } q;
#pragma unroll
  for (int p = 0; p < 8; ++p)
    q.u[p] = T[row][((p >> 2) << 3) + (lg << 2) + (p & 3)];
  return q.v;
}

__device__ inline v8f wmma_acc(v16bf a, v16bf b, v8f c) {
  return __builtin_amdgcn_wmma_f32_16x16x32_bf16(false, a, false, b, (short)0, c,
                                                 false, false);
}

// ------------------------------------------------------------ prep kernels ---

__device__ inline float mix_coeff(int k, int c, const float* mu, const float* sigma,
                                  float bias) {
  if (k < ND)      return -0.5f / sigma[k * NM + c];
  if (k < 2 * ND)  { int d = k - ND; return mu[d * NM + c] / sigma[d * NM + c]; }
  if (k == 2 * ND) return bias;
  return 0.f;
}

// Per-mixture coefficients for the logit GEMM, packed bf16 hi/lo pairs (NM x 64 pairs).
__global__ __launch_bounds__(256) void prep_mix(const float* __restrict__ mu,
                                                const float* __restrict__ sigma,
                                                const float* __restrict__ w,
                                                unsigned* __restrict__ Wph,
                                                unsigned* __restrict__ Wpl) {
  int c = blockIdx.x * blockDim.x + threadIdx.x;
  if (c >= NM) return;
  float logdet = 0.f, mm = 0.f;
  for (int d = 0; d < ND; ++d) {
    float sg = sigma[d * NM + c];
    float m  = mu[d * NM + c];
    logdet += __logf(sg);
    mm += m * m / sg;
  }
  const float LOG2PI = 1.8378770664093453f;
  float C    = mm + logdet - 2.f * __logf(w[c]);
  float bias = -0.5f * (C + (float)ND * LOG2PI);
  for (int kp = 0; kp < KF / 2; ++kp) {
    float v0 = mix_coeff(2 * kp,     c, mu, sigma, bias);
    float v1 = mix_coeff(2 * kp + 1, c, mu, sigma, bias);
    unsigned hi, lo;
    split2(v0, v1, hi, lo);
    Wph[c * 64 + kp] = hi;
    Wpl[c * 64 + kp] = lo;
  }
}

__device__ inline float feat_val(int k, const float* xr, int valid) {
  if (k < ND)      return xr[k] * xr[k];
  if (k < 2 * ND)  return xr[k - ND];
  if (k == 2 * ND) return valid ? 1.f : 0.f;
  return 0.f;
}

// Frame features [x^2, x, 1] as packed bf16 hi/lo (64 pairs x TP), plus Xaug (64 x TP):
// rows 0..59 = x^T, row 60 = ones (so the stats GEMM also produces N).
__global__ __launch_bounds__(256) void prep_frames(const float* __restrict__ x,
                                                   unsigned* __restrict__ Fph,
                                                   unsigned* __restrict__ Fpl,
                                                   float* __restrict__ Xaug) {
  __shared__ float xs[256][ND + 1];
  int t0 = blockIdx.x * 256;
  int tid = threadIdx.x;
  for (int i = tid; i < 256 * ND; i += 256) {   // linear -> coalesced
    int tt = i / ND, d = i % ND;
    int t = t0 + tt;
    xs[tt][d] = (t < NF) ? x[(size_t)t * ND + d] : 0.f;
  }
  __syncthreads();
  int t = t0 + tid;
  int valid = (t < NF);
  for (int d = 0; d < ND; ++d) Xaug[(size_t)d * TP + t] = xs[tid][d];
  Xaug[(size_t)ND * TP + t] = valid ? 1.f : 0.f;
  for (int d = ND + 1; d < 64; ++d) Xaug[(size_t)d * TP + t] = 0.f;
  for (int kp = 0; kp < KF / 2; ++kp) {
    float f0 = feat_val(2 * kp,     xs[tid], valid);
    float f1 = feat_val(2 * kp + 1, xs[tid], valid);
    unsigned hi, lo;
    split2(f0, f1, hi, lo);
    Fph[(size_t)kp * TP + t] = hi;
    Fpl[(size_t)kp * TP + t] = lo;
  }
}

// -------------------------------------------------- logits GEMM (bf16 WMMA) ---
// L[t, c] = W(2048 x 128) @ Feat(128 x 10240), split-bf16 x3, frame-major output.
__global__ __launch_bounds__(256) void logit_gemm(const unsigned* __restrict__ Wph,
                                                  const unsigned* __restrict__ Wpl,
                                                  const unsigned* __restrict__ Fph,
                                                  const unsigned* __restrict__ Fpl,
                                                  float* __restrict__ Lbuf) {
  __shared__ unsigned Ah[128][16], Al[128][16], Bh[128][16], Bl[128][16];
  int c0 = blockIdx.x * 128;
  int t0 = blockIdx.y * 128;
  int tid = threadIdx.x;
  int lane = tid & 31, wave = tid >> 5;
  int ln = lane & 15, lg = lane >> 4;
  int wm0 = (wave & 3) * 32;   // 2 m-tiles per wave
  int wn0 = (wave >> 2) * 64;  // 4 n-tiles per wave
  v8f acc[2][4];
#pragma unroll
  for (int a = 0; a < 2; ++a)
#pragma unroll
    for (int b = 0; b < 4; ++b) acc[a][b] = v8zero();

  for (int ch = 0; ch < KF / 32; ++ch) {
    int kp0 = ch * 16;
    __syncthreads();
    for (int i = tid; i < 2048; i += 256) {
      int m = i >> 4, kp = i & 15;
      Ah[m][kp] = Wph[(c0 + m) * 64 + kp0 + kp];
      Al[m][kp] = Wpl[(c0 + m) * 64 + kp0 + kp];
    }
    for (int i = tid; i < 2048; i += 256) {
      int kp = i >> 7, n = i & 127;
      Bh[n][kp] = Fph[(size_t)(kp0 + kp) * TP + t0 + n];
      Bl[n][kp] = Fpl[(size_t)(kp0 + kp) * TP + t0 + n];
    }
    __syncthreads();
    v16bf afh[2], afl[2];
#pragma unroll
    for (int mt = 0; mt < 2; ++mt) {
      afh[mt] = ldfrag(Ah, wm0 + mt * 16 + ln, lg);
      afl[mt] = ldfrag(Al, wm0 + mt * 16 + ln, lg);
    }
#pragma unroll
    for (int nt = 0; nt < 4; ++nt) {
      v16bf bfh = ldfrag(Bh, wn0 + nt * 16 + ln, lg);
      v16bf bfl = ldfrag(Bl, wn0 + nt * 16 + ln, lg);
#pragma unroll
      for (int mt = 0; mt < 2; ++mt) {
        acc[mt][nt] = wmma_acc(afh[mt], bfh, acc[mt][nt]);
        acc[mt][nt] = wmma_acc(afh[mt], bfl, acc[mt][nt]);
        acc[mt][nt] = wmma_acc(afl[mt], bfh, acc[mt][nt]);
      }
    }
  }
#pragma unroll
  for (int mt = 0; mt < 2; ++mt)
#pragma unroll
    for (int nt = 0; nt < 4; ++nt) {
      size_t base = (size_t)(t0 + wn0 + nt * 16 + ln) * NM +
                    (c0 + wm0 + mt * 16 + lg * 8);
#pragma unroll
      for (int v = 0; v < 8; ++v) Lbuf[base + v] = acc[mt][nt][v];
    }
}

// ------------------------------------------------------- softmax over mixes ---
__global__ __launch_bounds__(256) void softmax_rows(float* __restrict__ Lbuf) {
  __shared__ float red[256];
  int t = blockIdx.x;
  int tid = threadIdx.x;
  float* row = Lbuf + (size_t)t * NM;
  if (t >= NF) {
    for (int c = tid; c < NM; c += 256) row[c] = 0.f;
    return;
  }
  float r[8];
#pragma unroll
  for (int j = 0; j < 8; ++j) r[j] = row[tid + j * 256];
  float m = r[0];
#pragma unroll
  for (int j = 1; j < 8; ++j) m = fmaxf(m, r[j]);
  red[tid] = m;
  __syncthreads();
  for (int s = 128; s > 0; s >>= 1) {
    if (tid < s) red[tid] = fmaxf(red[tid], red[tid + s]);
    __syncthreads();
  }
  m = red[0];
  __syncthreads();
  float sum = 0.f;
#pragma unroll
  for (int j = 0; j < 8; ++j) { r[j] = __expf(r[j] - m); sum += r[j]; }
  red[tid] = sum;
  __syncthreads();
  for (int s = 128; s > 0; s >>= 1) {
    if (tid < s) red[tid] += red[tid + s];
    __syncthreads();
  }
  float inv = 1.f / red[0];
#pragma unroll
  for (int j = 0; j < 8; ++j) row[tid + j * 256] = r[j] * inv;
}

// -------------------------------- stats GEMM: Faug(64 x NM) = Xaug @ post ----
__global__ __launch_bounds__(256) void fgemm(const float* __restrict__ Xaug,
                                             const float* __restrict__ P,
                                             float* __restrict__ Faug) {
  __shared__ float Xs[64][17], Ps[16][65];
  int c0 = blockIdx.x * 64;
  int k0 = blockIdx.y * (TP / 16);   // 640-wide K slices, atomically reduced
  int tid = threadIdx.x, ty = tid >> 4, tx = tid & 15;
  float acc[4][4] = {{0.f}};
  for (int kk = 0; kk < TP / 16; kk += 16) {
    __syncthreads();
    for (int i = tid; i < 64 * 16; i += 256) {
      int d = i >> 4, j = i & 15;
      Xs[d][j] = Xaug[(size_t)d * TP + k0 + kk + j];
    }
    for (int i = tid; i < 16 * 64; i += 256) {
      int j = i >> 6, c = i & 63;
      Ps[j][c] = P[(size_t)(k0 + kk + j) * NM + c0 + c];
    }
    __syncthreads();
#pragma unroll 4
    for (int j = 0; j < 16; ++j) {
      float xa[4], pb[4];
#pragma unroll
      for (int a = 0; a < 4; ++a) xa[a] = Xs[ty * 4 + a][j];
#pragma unroll
      for (int b = 0; b < 4; ++b) pb[b] = Ps[j][tx * 4 + b];
#pragma unroll
      for (int a = 0; a < 4; ++a)
#pragma unroll
        for (int b = 0; b < 4; ++b) acc[a][b] += xa[a] * pb[b];
    }
  }
#pragma unroll
  for (int a = 0; a < 4; ++a)
#pragma unroll
    for (int b = 0; b < 4; ++b)
      atomicAdd(&Faug[(ty * 4 + a) * NM + c0 + tx * 4 + b], acc[a][b]);
}

// F_hat/Sigma and sqrt(N_c/Sigma) per flattened k = c*60+d (mix-major).
__global__ __launch_bounds__(256) void stats_fixup(const float* __restrict__ Faug,
                                                   const float* __restrict__ mu,
                                                   const float* __restrict__ sigma,
                                                   float* __restrict__ svec,
                                                   float* __restrict__ fhs) {
  int k = blockIdx.x * blockDim.x + threadIdx.x;
  if (k >= NK) return;
  int d = k % ND, c = k / ND;
  float N  = Faug[ND * NM + c];     // ones-row of the stats GEMM
  float F  = Faug[d * NM + c];
  float m  = mu[d * NM + c];
  float sg = sigma[d * NM + c];
  svec[k] = sqrtf(fmaxf(N, 0.f) / sg);
  fhs[k]  = (F - m * N) / sg;
}

// b[r] = Tm[r,:] . fhs  (re-reads Tm once: 295 MB ~ 12.7 us @ 23.3 TB/s)
__global__ __launch_bounds__(256) void bgemv(const float* __restrict__ Tm,
                                             const float* __restrict__ fhs,
                                             float* __restrict__ bvec) {
  __shared__ float red[256];
  int r = blockIdx.x;
  int tid = threadIdx.x;
  const float* rowp = Tm + (size_t)r * NK;
  float s = 0.f;
  for (int k = tid; k < NK; k += 256) s += rowp[k] * fhs[k];
  red[tid] = s;
  __syncthreads();
  for (int st = 128; st > 0; st >>= 1) {
    if (tid < st) red[tid] += red[tid + st];
    __syncthreads();
  }
  if (tid == 0) bvec[r] = red[0];
}

// -------- pack G = Tm * diag(svec) once into bf16 hi/lo packed k-pairs -------
// Removes the 20x-redundant in-loop f32->bf16 conversion from the SYRK.
__global__ __launch_bounds__(256) void pack_G(const float* __restrict__ Tm,
                                              const float* __restrict__ svec,
                                              unsigned* __restrict__ Ghp,
                                              unsigned* __restrict__ Glp) {
  int kp  = blockIdx.x * 256 + threadIdx.x;   // < NKP
  int row = blockIdx.y;                       // < RP
  float v0 = 0.f, v1 = 0.f;
  if (row < TV) {
    const float* rp = Tm + (size_t)row * NK + 2 * kp;
    v0 = rp[0] * svec[2 * kp];
    v1 = rp[1] * svec[2 * kp + 1];
  }
  unsigned hi, lo;
  split2(v0, v1, hi, lo);
  Ghp[(size_t)row * NKP + kp] = hi;
  Glp[(size_t)row * NKP + kp] = lo;
}

// ---- SYRK: upper-tri 128x128 tiles of M += G G^T, split-bf16 WMMA, ----------
// ---- double-buffered async global->LDS staging (ASYNCcnt pipeline) ----------
__global__ __launch_bounds__(256) void syrk_kernel(const unsigned* __restrict__ Ghp,
                                                   const unsigned* __restrict__ Glp,
                                                   float* __restrict__ Mmat) {
  // [buffer][slot: Ah,Al,Bh,Bl][row][kpair] = 64 KB
  __shared__ unsigned stage[2][4][128][16];

  // decode upper-triangular tile pair (ti <= tj) from blockIdx.x in [0, NT*(NT+1)/2)
  int p = blockIdx.x, ti = 0, rem = NT;
  while (p >= rem) { p -= rem; ++ti; --rem; }
  int tj = ti + p;
  int i0 = ti * 128, j0 = tj * 128;

  const int CH = (NK / 32) / KSPLIT;          // 240 K-chunks per slice
  int chunk0 = blockIdx.y * CH;

  int tid = threadIdx.x;
  int lane = tid & 31, wave = tid >> 5;
  int ln = lane & 15, lg = lane >> 4;
  int wm0 = (wave & 3) * 32;                  // 2 m-tiles per wave
  int wn0 = (wave >> 2) * 64;                 // 4 n-tiles per wave

  // per-thread fixed staging assignment: 8 x b128 async copies per chunk
  // item = w*256+tid -> slot(2 bits), row(7 bits), grp(2 bits of 4 kpairs)
  auto stage_load = [&](int buf, int chunk) {
    int kp0 = chunk * 16;
#pragma unroll
    for (int wI = 0; wI < 8; ++wI) {
      int item = wI * 256 + tid;
      int slot = item >> 9;
      int r    = (item >> 2) & 127;
      int grp  = item & 3;
      const unsigned* src = (slot & 1) ? Glp : Ghp;
      int rowbase = (slot < 2) ? i0 : j0;
      const unsigned* gp = src + (size_t)(rowbase + r) * NKP + kp0 + grp * 4;
      unsigned lds = (unsigned)(uintptr_t)&stage[buf][slot][r][grp * 4];
      unsigned long long ga = (unsigned long long)(uintptr_t)gp;
      asm volatile("global_load_async_to_lds_b128 %0, %1, off"
                   :: "v"(lds), "v"(ga) : "memory");
    }
  };

  v8f acc[2][4];
#pragma unroll
  for (int a = 0; a < 2; ++a)
#pragma unroll
    for (int b = 0; b < 4; ++b) acc[a][b] = v8zero();

  stage_load(0, chunk0);                      // prologue
  for (int cc = 0; cc < CH; ++cc) {
    int cur = cc & 1;
    if (cc + 1 < CH) {
      stage_load(cur ^ 1, chunk0 + cc + 1);   // prefetch next chunk
      asm volatile("s_wait_asynccnt 0x8" ::: "memory");  // current buffer landed
    } else {
      asm volatile("s_wait_asynccnt 0x0" ::: "memory");
    }
    __syncthreads();                          // all waves' staging visible

    const unsigned (*Ah)[16] = stage[cur][0];
    const unsigned (*Al)[16] = stage[cur][1];
    const unsigned (*Bh)[16] = stage[cur][2];
    const unsigned (*Bl)[16] = stage[cur][3];

    v16bf afh[2], afl[2];
#pragma unroll
    for (int mt = 0; mt < 2; ++mt) {
      afh[mt] = ldfrag(Ah, wm0 + mt * 16 + ln, lg);
      afl[mt] = ldfrag(Al, wm0 + mt * 16 + ln, lg);
    }
#pragma unroll
    for (int nt = 0; nt < 4; ++nt) {
      v16bf bfh = ldfrag(Bh, wn0 + nt * 16 + ln, lg);
      v16bf bfl = ldfrag(Bl, wn0 + nt * 16 + ln, lg);
#pragma unroll
      for (int mt = 0; mt < 2; ++mt) {
        acc[mt][nt] = wmma_acc(afh[mt], bfh, acc[mt][nt]);
        acc[mt][nt] = wmma_acc(afh[mt], bfl, acc[mt][nt]);
        acc[mt][nt] = wmma_acc(afl[mt], bfh, acc[mt][nt]);
      }
    }
    __syncthreads();                          // done reading before refill
  }

#pragma unroll
  for (int mt = 0; mt < 2; ++mt)
#pragma unroll
    for (int nt = 0; nt < 4; ++nt) {
      int rbase = i0 + wm0 + mt * 16 + lg * 8;
      int col   = j0 + wn0 + nt * 16 + ln;
#pragma unroll
      for (int v = 0; v < 8; ++v)
        atomicAdd(&Mmat[(size_t)(rbase + v) * RP + col], acc[mt][nt][v]);
    }
}

// lower triangle = transpose of computed upper triangle
__global__ __launch_bounds__(256) void mirror_lower(float* __restrict__ Mmat) {
  size_t idx = (size_t)blockIdx.x * 256 + threadIdx.x;
  int r = (int)(idx / RP), c = (int)(idx % RP);
  if (r > c) Mmat[idx] = Mmat[(size_t)c * RP + r];
}

__global__ void add_identity(float* __restrict__ Mmat) {
  int i = blockIdx.x * blockDim.x + threadIdx.x;
  if (i < RP) Mmat[(size_t)i * RP + i] += 1.f;
}

// ------------------------------ blocked LU (no pivot; M = I + PSD is safe) ---
__global__ __launch_bounds__(256) void lu_panel(float* __restrict__ A, int kb) {
  __shared__ float D[64][65];
  int tid = threadIdx.x;
  int base = kb * 64;
  for (int i = tid; i < 64 * 64; i += 256)
    D[i >> 6][i & 63] = A[(size_t)(base + (i >> 6)) * RP + base + (i & 63)];
  __syncthreads();
  for (int j = 0; j < 64; ++j) {
    if (tid > j && tid < 64) {
      float lij = D[tid][j] / D[j][j];
      D[tid][j] = lij;
      for (int p = j + 1; p < 64; ++p) D[tid][p] -= lij * D[j][p];
    }
    __syncthreads();
  }
  for (int i = tid; i < 64 * 64; i += 256)
    A[(size_t)(base + (i >> 6)) * RP + base + (i & 63)] = D[i >> 6][i & 63];
  int rem = RP - base - 64;
  // column panel: X * U_kk = A  (rows below)
  for (int i = tid; i < rem; i += 256) {
    int row = base + 64 + i;
    float xr[64];
    for (int j = 0; j < 64; ++j) xr[j] = A[(size_t)row * RP + base + j];
    for (int j = 0; j < 64; ++j) {
      float xj = xr[j] / D[j][j];
      xr[j] = xj;
      for (int p = j + 1; p < 64; ++p) xr[p] -= xj * D[j][p];
    }
    for (int j = 0; j < 64; ++j) A[(size_t)row * RP + base + j] = xr[j];
  }
  // row panel: L_kk^{-1} * A  (cols right), L unit-diag
  for (int i = tid; i < rem; i += 256) {
    int col = base + 64 + i;
    float yr[64];
    for (int j = 0; j < 64; ++j) yr[j] = A[(size_t)(base + j) * RP + col];
    for (int j = 0; j < 64; ++j) {
      float yj = yr[j];
      for (int p = j + 1; p < 64; ++p) yr[p] -= D[p][j] * yj;
    }
    for (int j = 0; j < 64; ++j) A[(size_t)(base + j) * RP + col] = yr[j];
  }
}

__global__ __launch_bounds__(256) void lu_update(float* __restrict__ A, int kb) {
  __shared__ float Ls[64][17], Us[16][65];
  int base = kb * 64, off = base + 64;
  int i0 = off + blockIdx.y * 64, j0 = off + blockIdx.x * 64;
  int tid = threadIdx.x, ty = tid >> 4, tx = tid & 15;
  float acc[4][4] = {{0.f}};
  for (int kk = 0; kk < 64; kk += 16) {
    __syncthreads();
    for (int i = tid; i < 64 * 16; i += 256) {
      int r = i >> 4, c = i & 15;
      Ls[r][c] = A[(size_t)(i0 + r) * RP + base + kk + c];
    }
    for (int i = tid; i < 16 * 64; i += 256) {
      int r = i >> 6, c = i & 63;
      Us[r][c] = A[(size_t)(base + kk + r) * RP + j0 + c];
    }
    __syncthreads();
#pragma unroll 4
    for (int j = 0; j < 16; ++j) {
      float xa[4], ub[4];
#pragma unroll
      for (int a = 0; a < 4; ++a) xa[a] = Ls[ty * 4 + a][j];
#pragma unroll
      for (int b = 0; b < 4; ++b) ub[b] = Us[j][tx * 4 + b];
#pragma unroll
      for (int a = 0; a < 4; ++a)
#pragma unroll
        for (int b = 0; b < 4; ++b) acc[a][b] += xa[a] * ub[b];
    }
  }
#pragma unroll
  for (int a = 0; a < 4; ++a)
#pragma unroll
    for (int b = 0; b < 4; ++b)
      A[(size_t)(i0 + ty * 4 + a) * RP + j0 + tx * 4 + b] -= acc[a][b];
}

__global__ __launch_bounds__(256) void lu_solve(const float* __restrict__ A,
                                                const float* __restrict__ b,
                                                float* __restrict__ out) {
  __shared__ float y[RP];
  __shared__ float red[256];
  int tid = threadIdx.x;
  for (int k = 0; k < RP; ++k) {              // L y = b (unit lower)
    float s = 0.f;
    for (int j = tid; j < k; j += 256) s += A[(size_t)k * RP + j] * y[j];
    red[tid] = s;
    __syncthreads();
    for (int st = 128; st > 0; st >>= 1) {
      if (tid < st) red[tid] += red[tid + st];
      __syncthreads();
    }
    if (tid == 0) y[k] = b[k] - red[0];
    __syncthreads();
  }
  for (int k = RP - 1; k >= 0; --k) {         // U x = y (in place)
    float s = 0.f;
    for (int j = k + 1 + tid; j < RP; j += 256) s += A[(size_t)k * RP + j] * y[j];
    red[tid] = s;
    __syncthreads();
    for (int st = 128; st > 0; st >>= 1) {
      if (tid < st) red[tid] += red[tid + st];
      __syncthreads();
    }
    if (tid == 0) y[k] = (y[k] - red[0]) / A[(size_t)k * RP + k];
    __syncthreads();
  }
  for (int i = tid; i < TV; i += 256) out[i] = y[i];
}

// ------------------------------------------------------------------- host ---

extern "C" void kernel_launch(void* const* d_in, const int* in_sizes, int n_in,
                              void* d_out, int out_size, void* d_ws, size_t ws_size,
                              hipStream_t stream) {
  const float* x     = (const float*)d_in[0];
  const float* mu    = (const float*)d_in[1];
  const float* sigma = (const float*)d_in[2];
  const float* w     = (const float*)d_in[3];
  const float* Tm    = (const float*)d_in[4];
  float* out = (float*)d_out;

  char* base = (char*)d_ws;
  size_t off = 0;
  auto carve = [&](size_t bytes) -> void* {
    void* r = base + off;
    off += (bytes + 255) & ~(size_t)255;
    return r;
  };
  unsigned* Wph  = (unsigned*)carve((size_t)NM * 64 * 4);
  unsigned* Wpl  = (unsigned*)carve((size_t)NM * 64 * 4);
  unsigned* Fph  = (unsigned*)carve((size_t)64 * TP * 4);
  unsigned* Fpl  = (unsigned*)carve((size_t)64 * TP * 4);
  float*    Lbuf = (float*)carve((size_t)TP * NM * 4);
  float*    Xaug = (float*)carve((size_t)64 * TP * 4);
  float*    Faug = (float*)carve((size_t)64 * NM * 4);
  float*    svec = (float*)carve((size_t)NK * 4);
  float*    fhs  = (float*)carve((size_t)NK * 4);
  float*    Mmat = (float*)carve((size_t)RP * RP * 4);
  float*    bvec = (float*)carve((size_t)RP * 4);
  unsigned* Ghp  = (unsigned*)carve((size_t)RP * NKP * 4);
  unsigned* Glp  = (unsigned*)carve((size_t)RP * NKP * 4);
  (void)in_sizes; (void)n_in; (void)out_size; (void)ws_size;

  hipMemsetAsync(Faug, 0, (size_t)64 * NM * 4, stream);
  hipMemsetAsync(Mmat, 0, (size_t)RP * RP * 4, stream);
  hipMemsetAsync(bvec, 0, (size_t)RP * 4, stream);

  prep_mix<<<NM / 256, 256, 0, stream>>>(mu, sigma, w, Wph, Wpl);
  prep_frames<<<TP / 256, 256, 0, stream>>>(x, Fph, Fpl, Xaug);
  logit_gemm<<<dim3(NM / 128, TP / 128), 256, 0, stream>>>(Wph, Wpl, Fph, Fpl, Lbuf);
  softmax_rows<<<TP, 256, 0, stream>>>(Lbuf);
  fgemm<<<dim3(NM / 64, 16), 256, 0, stream>>>(Xaug, Lbuf, Faug);
  stats_fixup<<<NK / 256, 256, 0, stream>>>(Faug, mu, sigma, svec, fhs);
  bgemv<<<TV, 256, 0, stream>>>(Tm, fhs, bvec);
  pack_G<<<dim3(NKP / 256, RP), 256, 0, stream>>>(Tm, svec, Ghp, Glp);
  syrk_kernel<<<dim3(NT * (NT + 1) / 2, KSPLIT), 256, 0, stream>>>(Ghp, Glp, Mmat);
  mirror_lower<<<(RP * RP) / 256, 256, 0, stream>>>(Mmat);
  add_identity<<<(RP + 255) / 256, 256, 0, stream>>>(Mmat);
  for (int kb = 0; kb < RP / 64; ++kb) {
    lu_panel<<<1, 256, 0, stream>>>(Mmat, kb);
    int nt = RP / 64 - kb - 1;
    if (nt > 0) lu_update<<<dim3(nt, nt), 256, 0, stream>>>(Mmat, kb);
  }
  lu_solve<<<1, 256, 0, stream>>>(Mmat, bvec, out);
}
